// HET_HGTLayerHetero_13176959664631
// MI455X (gfx1250) — matrix operand order repaired and checked by reading, not compiled, not run on hardware
//
#include <hip/hip_runtime.h>

#define N_   131072
#define E_   393216
#define T_   4
#define R_   6
#define IN_  256
#define OUT_ 256
#define H_   4
#define DK_  64
#define NPT  (N_/T_)   // 32768 nodes per type
#define EPR  (E_/R_)   // 65536 edges per relation

typedef __attribute__((ext_vector_type(16))) __bf16 v16bf;
typedef __attribute__((ext_vector_type(8)))  __bf16 v8bf;
typedef __attribute__((ext_vector_type(8)))  float  v8f;
typedef __attribute__((ext_vector_type(4)))  float  v4f;

static __device__ __forceinline__ v8f wmma_bf16(v16bf a, v16bf b, v8f c) {
  // D(f32 16x16) = A(bf16 16x32) * B(bf16 32x16) + C
  return __builtin_amdgcn_wmma_f32_16x16x32_bf16(false, a, false, b, (short)0, c,
                                                 false, false);
}

// A fragment (16-bit A 16x32): lane holds row M = lane&15; half = lane>>4.
// Lane's 16 elems = K[half*16 .. +16)  (two contiguous 16B runs)
static __device__ __forceinline__ v16bf a_frag_bf16(const __bf16* row, int kbase, int half) {
  union { v16bf v; v8bf h[2]; } u;
  u.h[0] = *(const v8bf*)(row + kbase + half * 16);
  u.h[1] = *(const v8bf*)(row + kbase + half * 16 + 8);
  return u.v;
}

// Same A fragment, but source rows are f32 (convert to bf16 in-register).
static __device__ __forceinline__ v16bf a_frag_f32(const float* row, int kbase, int half) {
  union { v16bf v; __bf16 e[16]; } u;
  const v4f* p0 = (const v4f*)(row + kbase + half * 16);
  const v4f* p1 = (const v4f*)(row + kbase + half * 16 + 8);
  v4f f0 = p0[0], f1 = p0[1], f2 = p1[0], f3 = p1[1];
#pragma unroll
  for (int j = 0; j < 4; ++j) {
    u.e[j]      = (__bf16)f0[j];
    u.e[4 + j]  = (__bf16)f1[j];
    u.e[8 + j]  = (__bf16)f2[j];
    u.e[12 + j] = (__bf16)f3[j];
  }
  return u.v;
}

// B fragment (16-bit B 32x16): lane holds col N = lane&15, K = half*16 .. +16
// contiguous when weights are pre-transposed to [n][k] layout.
static __device__ __forceinline__ v16bf b_frag(const __bf16* p) {
  union { v16bf v; v8bf h[2]; } u;
  u.h[0] = ((const v8bf*)p)[0];
  u.h[1] = ((const v8bf*)p)[1];
  return u.v;
}

// ---------------------------------------------------------------------------
// K0: convert + transpose weights to bf16.
//   wt3  [3][T][OUT][IN]  <- {k_w,q_w,v_w}[T][IN][OUT]
//   wat  [T][OUT][OUT]    <- a_w[T][OUT][OUT] transposed
//   attT/msgT [R][H][DK][DK] <- relation_{att,msg}[R][H][DK][DK] transposed
// ---------------------------------------------------------------------------
__global__ void k_prep(const float* __restrict__ kw, const float* __restrict__ qw,
                       const float* __restrict__ vw, const float* __restrict__ aw,
                       const float* __restrict__ att, const float* __restrict__ msg,
                       __bf16* __restrict__ wt3, __bf16* __restrict__ wat,
                       __bf16* __restrict__ attT, __bf16* __restrict__ msgT) {
  int i = blockIdx.x * blockDim.x + threadIdx.x;
  const int WKQV = 3 * T_ * IN_ * OUT_;
  const int WA   = T_ * OUT_ * OUT_;
  const int WR   = R_ * H_ * DK_ * DK_;
  if (i < WKQV) {
    int m  = i / (T_ * IN_ * OUT_);
    int rem = i % (T_ * IN_ * OUT_);
    int t  = rem / (IN_ * OUT_);
    int rr = rem % (IN_ * OUT_);
    int o = rr / IN_, k = rr % IN_;
    const float* w = (m == 0) ? kw : ((m == 1) ? qw : vw);
    wt3[i] = (__bf16)w[((size_t)t * IN_ + k) * OUT_ + o];
  } else if (i < WKQV + WA) {
    int j = i - WKQV;
    int t = j / (OUT_ * OUT_);
    int rr = j % (OUT_ * OUT_);
    int o = rr / OUT_, k = rr % OUT_;
    wat[j] = (__bf16)aw[((size_t)t * OUT_ + k) * OUT_ + o];
  } else if (i < WKQV + WA + 2 * WR) {
    int j = i - WKQV - WA;
    int which = j / WR;
    int jj = j % WR;
    int rh = jj / (DK_ * DK_);
    int rr = jj % (DK_ * DK_);
    int d = rr / DK_, k = rr % DK_;
    const float* ten = which ? msg : att;
    __bf16* dp = which ? msgT : attT;
    dp[jj] = (__bf16)ten[((size_t)rh * DK_ + k) * DK_ + d];
  }
}

// K_init: zero softmax denominators, encoded segment-max, and agg (= d_out scratch)
__global__ void k_zero(float* __restrict__ den, unsigned* __restrict__ smax,
                       float* __restrict__ agg) {
  size_t i = (size_t)blockIdx.x * blockDim.x + threadIdx.x;
  if (i < (size_t)N_ * H_) { den[i] = 0.f; smax[i] = 0u; }
  if (i < (size_t)N_ * OUT_) agg[i] = 0.f;
}

// ---------------------------------------------------------------------------
// K1: per-type fused K/Q/V projection GEMM. grid = (N/16 strips, 3 matrices),
// block = 128 (4 waves x 64 output cols). bf16 output (L2-resident for edges).
// ---------------------------------------------------------------------------
__global__ void k_proj(const float* __restrict__ hsrc, const __bf16* __restrict__ wt3,
                       const float* __restrict__ bk, const float* __restrict__ bq,
                       const float* __restrict__ bv, __bf16* __restrict__ kqv) {
  const int node0 = blockIdx.x * 16;
  const int m = blockIdx.y;                 // 0=k 1=q 2=v
  const int t = node0 / NPT;
  const int lane = threadIdx.x & 31, wave = threadIdx.x >> 5;
  const int half = lane >> 4, col = lane & 15;
  const __bf16* wt = wt3 + ((size_t)m * T_ + t) * OUT_ * IN_;
  const float* bias = ((m == 0) ? bk : (m == 1) ? bq : bv) + (size_t)t * OUT_;
  __bf16* out = kqv + (size_t)m * N_ * OUT_;
  const float* arow = hsrc + (size_t)(node0 + col) * IN_;
  v8f c[4] = {};
  for (int kk = 0; kk < IN_ / 32; ++kk) {
    v16bf a = a_frag_f32(arow, kk * 32, half);
#pragma unroll
    for (int tt = 0; tt < 4; ++tt) {
      int ncol = wave * 64 + tt * 16 + col;
      c[tt] = wmma_bf16(a, b_frag(wt + (size_t)ncol * IN_ + kk * 32 + half * 16), c[tt]);
    }
  }
#pragma unroll
  for (int tt = 0; tt < 4; ++tt) {
    int ncol = wave * 64 + tt * 16 + col;
    float b = bias[ncol];
#pragma unroll
    for (int r = 0; r < 8; ++r)
      out[(size_t)(node0 + r + half * 8) * OUT_ + ncol] = (__bf16)(c[tt][r] + b);
  }
}

// ---------------------------------------------------------------------------
// K2: edge scores. One wave per 16-edge group (groups never straddle a relation).
// qw = q[dst] @ att^T via WMMA; dot with LDS-staged k[src]; butterfly-reduce;
// segment-max via order-preserving uint atomicMax.
// ---------------------------------------------------------------------------
__global__ void k_score(const __bf16* __restrict__ kb, const __bf16* __restrict__ qb,
                        const int* __restrict__ src, const int* __restrict__ dst,
                        const __bf16* __restrict__ attT, const float* __restrict__ pri,
                        float* __restrict__ score, unsigned* __restrict__ smax) {
  __shared__ __bf16 kst[8][16][DK_];        // per-wave 16 edges x one head slice
  const int lane = threadIdx.x & 31, wave = threadIdx.x >> 5;
  const int half = lane >> 4, col = lane & 15;
  const int ebase = (blockIdx.x * 8 + wave) * 16;
  const int rel = ebase / EPR;
  const int myedge = ebase + col;
  const __bf16* krow = kb + (size_t)src[myedge] * OUT_;
  const __bf16* qrow = qb + (size_t)dst[myedge] * OUT_;
  for (int hh = 0; hh < H_; ++hh) {
    { // stage k[src] head slice into LDS (each lane: 32 contiguous bf16)
      const v8bf* ps = (const v8bf*)(krow + hh * 64 + half * 32);
      v8bf* pd = (v8bf*)&kst[wave][col][half * 32];
#pragma unroll
      for (int j = 0; j < 4; ++j) pd[j] = ps[j];
    }
    __syncthreads();
    v8f c[4] = {};
#pragma unroll
    for (int kk = 0; kk < 2; ++kk) {
      v16bf a = a_frag_bf16(qrow, hh * 64 + kk * 32, half);
#pragma unroll
      for (int tt = 0; tt < 4; ++tt) {
        const __bf16* bp = attT + (((size_t)rel * H_ + hh) * DK_ + tt * 16 + col) * DK_
                                + kk * 32 + half * 16;
        c[tt] = wmma_bf16(a, b_frag(bp), c[tt]);
      }
    }
    float p[8];
#pragma unroll
    for (int r = 0; r < 8; ++r) {
      int row = r + half * 8;
      float acc = 0.f;
#pragma unroll
      for (int tt = 0; tt < 4; ++tt)
        acc += c[tt][r] * (float)kst[wave][row][tt * 16 + col];
      p[r] = acc;
    }
#pragma unroll
    for (int r = 0; r < 8; ++r)
#pragma unroll
      for (int m = 1; m < 16; m <<= 1)
        p[r] += __shfl_xor(p[r], m, 32);
    if (col == 0) {
      const float scale = pri[rel * H_ + hh] * 0.125f;   // pri / sqrt(64)
#pragma unroll
      for (int r = 0; r < 8; ++r) {
        int e = ebase + r + half * 8;
        float sc = p[r] * scale;
        score[(size_t)e * H_ + hh] = sc;
        unsigned u = __float_as_uint(sc);
        unsigned enc = (u & 0x80000000u) ? ~u : (u | 0x80000000u);
        atomicMax(&smax[(size_t)dst[e] * H_ + hh], enc);
      }
    }
    __syncthreads();
  }
}

// K3: exp(score - max) and denominator segment-sum
__global__ void k_soft(float* __restrict__ score, const int* __restrict__ dst,
                       const unsigned* __restrict__ smax, float* __restrict__ den) {
  size_t i = (size_t)blockIdx.x * blockDim.x + threadIdx.x;
  if (i >= (size_t)E_ * H_) return;
  int e = (int)(i / H_), hh = (int)(i % H_);
  int d = dst[e];
  unsigned enc = smax[(size_t)d * H_ + hh];
  unsigned u = (enc & 0x80000000u) ? (enc & 0x7FFFFFFFu) : ~enc;
  float ex = __expf(score[i] - __uint_as_float(u));
  score[i] = ex;
  atomicAdd(&den[(size_t)d * H_ + hh], ex);
}

// ---------------------------------------------------------------------------
// K4: recompute msg = v[src] @ msg^T via WMMA, weight by softmax, scatter-add
// into agg (d_out) straight from the C-fragment layout (lane = (edge,dk)).
// ---------------------------------------------------------------------------
__global__ void k_agg(const __bf16* __restrict__ vb, const int* __restrict__ src,
                      const int* __restrict__ dst, const __bf16* __restrict__ msgT,
                      const float* __restrict__ score, const float* __restrict__ den,
                      float* __restrict__ agg) {
  const int lane = threadIdx.x & 31, wave = threadIdx.x >> 5;
  const int half = lane >> 4, col = lane & 15;
  const int ebase = (blockIdx.x * 8 + wave) * 16;
  const int rel = ebase / EPR;
  const __bf16* vrow = vb + (size_t)src[ebase + col] * OUT_;
  for (int hh = 0; hh < H_; ++hh) {
    v8f c[4] = {};
#pragma unroll
    for (int kk = 0; kk < 2; ++kk) {
      v16bf a = a_frag_bf16(vrow, hh * 64 + kk * 32, half);
#pragma unroll
      for (int tt = 0; tt < 4; ++tt) {
        const __bf16* bp = msgT + (((size_t)rel * H_ + hh) * DK_ + tt * 16 + col) * DK_
                                + kk * 32 + half * 16;
        c[tt] = wmma_bf16(a, b_frag(bp), c[tt]);
      }
    }
    int dsts[8]; float w[8];
#pragma unroll
    for (int r = 0; r < 8; ++r) {
      int e = ebase + r + half * 8;
      int d = dst[e];
      dsts[r] = d;
      w[r] = score[(size_t)e * H_ + hh] / den[(size_t)d * H_ + hh];
    }
#pragma unroll
    for (int tt = 0; tt < 4; ++tt)
#pragma unroll
      for (int r = 0; r < 8; ++r)
        atomicAdd(&agg[(size_t)dsts[r] * OUT_ + hh * 64 + tt * 16 + col],
                  w[r] * c[tt][r]);
  }
}

// ---------------------------------------------------------------------------
// K5: output projection, in place on d_out. Each block owns a 16-row strip;
// barrier between the K-loop reads and the writes makes in-place safe.
// ---------------------------------------------------------------------------
__global__ void k_out(float* __restrict__ out, const __bf16* __restrict__ wat,
                      const float* __restrict__ ab, const float* __restrict__ skip) {
  const int node0 = blockIdx.x * 16;
  const int t = node0 / NPT;
  const int lane = threadIdx.x & 31, wave = threadIdx.x >> 5;
  const int half = lane >> 4, col = lane & 15;
  const __bf16* wt = wat + (size_t)t * OUT_ * OUT_;
  const float* arow = out + (size_t)(node0 + col) * OUT_;
  v8f c[4] = {};
  for (int kk = 0; kk < OUT_ / 32; ++kk) {
    v16bf a = a_frag_f32(arow, kk * 32, half);
#pragma unroll
    for (int tt = 0; tt < 4; ++tt) {
      int ncol = wave * 64 + tt * 16 + col;
      c[tt] = wmma_bf16(a, b_frag(wt + (size_t)ncol * OUT_ + kk * 32 + half * 16), c[tt]);
    }
  }
  const float alpha = 1.f / (1.f + __expf(-skip[t]));
  __syncthreads();   // all strip reads consumed before in-place writes
#pragma unroll
  for (int tt = 0; tt < 4; ++tt) {
    int ncol = wave * 64 + tt * 16 + col;
    float b = ab[(size_t)t * OUT_ + ncol];
#pragma unroll
    for (int r = 0; r < 8; ++r)
      out[(size_t)(node0 + r + half * 8) * OUT_ + ncol] = (c[tt][r] + b) * alpha;
  }
}

extern "C" void kernel_launch(void* const* d_in, const int* in_sizes, int n_in,
                              void* d_out, int out_size, void* d_ws, size_t ws_size,
                              hipStream_t stream) {
  const float* h     = (const float*)d_in[0];
  const int*   src   = (const int*)d_in[1];
  const int*   dst   = (const int*)d_in[2];
  const float* kw    = (const float*)d_in[3];
  const float* kb    = (const float*)d_in[4];
  const float* qw    = (const float*)d_in[5];
  const float* qb    = (const float*)d_in[6];
  const float* vw    = (const float*)d_in[7];
  const float* vb    = (const float*)d_in[8];
  const float* aw    = (const float*)d_in[9];
  const float* ab    = (const float*)d_in[10];
  const float* pri   = (const float*)d_in[11];
  const float* att   = (const float*)d_in[12];
  const float* msg   = (const float*)d_in[13];
  const float* skip  = (const float*)d_in[14];
  float* out = (float*)d_out;

  char* ws = (char*)d_ws;
  size_t off = 0;
  __bf16* kqv  = (__bf16*)(ws + off); off += (size_t)3 * N_ * OUT_ * 2;    // 192 MB
  __bf16* wt3  = (__bf16*)(ws + off); off += (size_t)3 * T_ * IN_ * OUT_ * 2;
  __bf16* wat  = (__bf16*)(ws + off); off += (size_t)T_ * OUT_ * OUT_ * 2;
  __bf16* attT = (__bf16*)(ws + off); off += (size_t)R_ * H_ * DK_ * DK_ * 2;
  __bf16* msgT = (__bf16*)(ws + off); off += (size_t)R_ * H_ * DK_ * DK_ * 2;
  float*  scre = (float*)(ws + off);  off += (size_t)E_ * H_ * 4;          // 6 MB
  unsigned* smax = (unsigned*)(ws + off); off += (size_t)N_ * H_ * 4;
  float*  den  = (float*)(ws + off);  off += (size_t)N_ * H_ * 4;

  const int PREP = 3 * T_ * IN_ * OUT_ + T_ * OUT_ * OUT_ + 2 * R_ * H_ * DK_ * DK_;
  k_prep<<<(PREP + 255) / 256, 256, 0, stream>>>(kw, qw, vw, aw, att, msg,
                                                 wt3, wat, attT, msgT);
  k_zero<<<(N_ * OUT_) / 256, 256, 0, stream>>>(den, smax, out);
  k_proj<<<dim3(N_ / 16, 3), 128, 0, stream>>>(h, wt3, kb, qb, vb, kqv);
  k_score<<<E_ / 128, 256, 0, stream>>>(kqv, kqv + (size_t)N_ * OUT_, src, dst,
                                        attT, pri, scre, smax);
  k_soft<<<(E_ * H_) / 256, 256, 0, stream>>>(scre, dst, smax, den);
  k_agg<<<E_ / 128, 256, 0, stream>>>(kqv + (size_t)2 * N_ * OUT_, src, dst,
                                      msgT, scre, den, out);
  k_out<<<N_ / 16, 128, 0, stream>>>(out, wat, ab, skip);
}
